// KpcaStd_43714177139045
// MI455X (gfx1250) — matrix operand order repaired and checked by reading, not compiled, not run on hardware
//
#include <hip/hip_runtime.h>

typedef __attribute__((ext_vector_type(16))) __bf16 v16bf;
typedef __attribute__((ext_vector_type(8)))  __bf16 v8bf;
typedef __attribute__((ext_vector_type(8)))  float  v8f;

#define N_SAMPLE 8192
#define DIM      512
#define S_DIM    64
#define INV_SIGMA2 (1.0f / 512.0f)

// workspace layout (bytes)
#define XB_OFF  0u                      // bf16 X       [8192][512]  = 8 MB
#define WT_OFF  (8u * 1024u * 1024u)    // bf16 W^T     [64][8192]   = 1 MB
#define SQ_OFF  (9u * 1024u * 1024u)    // f32 ||x||^2  [8192]       = 32 KB
#define ACC_OFF (SQ_OFF + 32u * 1024u)  // f32 loss accumulator

// LDS layout (bytes, within dynamic shared)
#define L_XBJ   0u        // [64][512] bf16, 64 KB resident J strip
#define L_XBI   65536u    // 2 x [32][512] bf16 (double buffer, stride 32 KB)
#define L_WTC   131072u   // 2 x [64][32] bf16  (double buffer, stride 4 KB)
#define L_KT    139264u   // [4][16][32] bf16 per-wave K^T scratch
#define L_TOTAL 143360u   // 140 KB of the 320 KB WGP LDS

union V16U { v16bf v; struct { v8bf lo, hi; } p; };

__device__ __forceinline__ v16bf ld2(const __bf16* p0, const __bf16* p1) {
  V16U u;
  u.p.lo = *(const v8bf*)p0;
  u.p.hi = *(const v8bf*)p1;
  return u.v;
}

// ---- CDNA5 async global->LDS copy (ASYNCcnt-tracked), GV addressing ----
__device__ __forceinline__ void async_b128(unsigned ldsOff, const void* g) {
  asm volatile("global_load_async_to_lds_b128 %0, %1, off"
               :: "v"(ldsOff), "v"(g) : "memory");
}
__device__ __forceinline__ void async_wait0() {
  asm volatile("s_wait_asynccnt 0" ::: "memory");
}

__device__ __forceinline__ void stage_xbi(unsigned ldsOff, const __bf16* Xb,
                                          int i0, int tid) {
  const char* g = (const char*)(Xb + (size_t)i0 * DIM);   // 32 rows x 512 bf16
  for (int t = tid; t < 2048; t += 128)
    async_b128(ldsOff + (unsigned)t * 16u, g + (size_t)t * 16u);
}
__device__ __forceinline__ void stage_wtc(unsigned ldsOff, const __bf16* Wt,
                                          int i0, int tid) {
  for (int t = tid; t < 256; t += 128) {                  // 64 rows x 32 bf16
    int s = t >> 2, q = t & 3;
    async_b128(ldsOff + (unsigned)t * 16u,
               (const char*)(Wt + (size_t)s * N_SAMPLE + i0) + q * 16);
  }
}

// ---------------------------------------------------------------------------
// prep: X -> bf16, W -> bf16 transposed, row norms, zero accumulator
// ---------------------------------------------------------------------------
__global__ __launch_bounds__(256) void kpca_prep(
    const float* __restrict__ x, const float* __restrict__ W,
    __bf16* __restrict__ Xb, __bf16* __restrict__ Wt,
    float* __restrict__ sq, float* __restrict__ accum)
{
  const int r = blockIdx.x;     // sample row
  const int t = threadIdx.x;    // 256 threads
  float v0 = x[(size_t)r * DIM + t];
  float v1 = x[(size_t)r * DIM + t + 256];
  Xb[(size_t)r * DIM + t]       = (__bf16)v0;
  Xb[(size_t)r * DIM + t + 256] = (__bf16)v1;

  __shared__ float red[256];
  red[t] = v0 * v0 + v1 * v1;
  __syncthreads();
  for (int o = 128; o > 0; o >>= 1) {
    if (t < o) red[t] += red[t + o];
    __syncthreads();
  }
  if (t == 0) sq[r] = red[0];

  if (t < S_DIM) Wt[(size_t)t * N_SAMPLE + r] = (__bf16)W[(size_t)r * S_DIM + t];
  if (r == 0 && t == 0) accum[0] = 0.0f;
}

// ---------------------------------------------------------------------------
// main: fused  K-tile (bf16 WMMA gram + exp)  ->  E += Wt * Kbf (bf16 WMMA)
// 128 WGs x 128 threads (4 waves); async double-buffered LDS staging.
// ---------------------------------------------------------------------------
__global__ __launch_bounds__(128) void kpca_main(
    const __bf16* __restrict__ Xb, const __bf16* __restrict__ Wt,
    const float* __restrict__ sq, const float* __restrict__ W,
    const float* __restrict__ invl, float* __restrict__ accum)
{
  extern __shared__ char smem[];
  const unsigned ldsBase = (unsigned)(size_t)smem;  // low 32 bits = LDS offset

  __bf16* sXbJ = (__bf16*)(smem + L_XBJ);
  __bf16* sKT  = (__bf16*)(smem + L_KT);

  const int tid  = threadIdx.x;
  const int wave = tid >> 5;
  const int lane = tid & 31;
  const int l16  = lane & 15;
  const int half = lane >> 4;
  const int j0   = blockIdx.x * 64;
  const int j    = j0 + wave * 16 + l16;

  // prologue: async-stage resident J strip + chunk 0 (double-buffer prime)
  {
    const char* g = (const char*)(Xb + (size_t)j0 * DIM);
    for (int t = tid; t < 4096; t += 128)
      async_b128(ldsBase + L_XBJ + (unsigned)t * 16u, g + (size_t)t * 16u);
  }
  stage_xbi(ldsBase + L_XBI, Xb, 0, tid);
  stage_wtc(ldsBase + L_WTC, Wt, 0, tid);
  async_wait0();
  __syncthreads();

  const float sqj = sq[j];
  const __bf16* bRow = sXbJ + (wave * 16 + l16) * DIM;
  __bf16* kTw = sKT + wave * (16 * 32);                  // [j_local][i_local]

  v8f accA[4] = {};                                      // E[s(64), j(16)] per wave

  for (int c = 0; c < N_SAMPLE / 32; ++c) {
    const int i0 = c * 32;
    const unsigned cb = (unsigned)(c & 1);

    // issue next chunk into the other buffer (safe: barrier at end of prev
    // iteration guarantees every wave is done reading it)
    if (c + 1 < N_SAMPLE / 32) {
      stage_xbi(ldsBase + L_XBI + (cb ^ 1u) * 32768u, Xb, i0 + 32, tid);
      stage_wtc(ldsBase + L_WTC + (cb ^ 1u) * 4096u,  Wt, i0 + 32, tid);
    }

    const __bf16* sXbIc = (const __bf16*)(smem + L_XBI + cb * 32768u);
    const __bf16* sWtCc = (const __bf16*)(smem + L_WTC + cb * 4096u);
    const __bf16* aRow0 = sXbIc + l16 * DIM;             // gram A tile 0 (i 0..15)
    const __bf16* aRow1 = sXbIc + (l16 + 16) * DIM;      // gram A tile 1 (i 16..31)

    // ---- Gram tiles: two 16x16 fp32 tiles over K = 512 via bf16 WMMA ----
    v8f d0 = {}, d1 = {};
    #pragma unroll
    for (int dk = 0; dk < DIM; dk += 32) {
      v16bf a0 = ld2(aRow0 + dk + half * 8, aRow0 + dk + 16 + half * 8);
      v16bf a1 = ld2(aRow1 + dk + half * 8, aRow1 + dk + 16 + half * 8);
      v16bf b  = ld2(bRow + dk + half * 8, bRow + dk + 16 + half * 8);
      d0 = __builtin_amdgcn_wmma_f32_16x16x32_bf16(false, a0, false, b, (short)0, d0, false, false);
      d1 = __builtin_amdgcn_wmma_f32_16x16x32_bf16(false, a1, false, b, (short)0, d1, false, false);
    }

    // ---- K = exp((g - 0.5(sq_i + sq_j))/sigma2), pack bf16, store j-major ----
    #pragma unroll
    for (int t2 = 0; t2 < 2; ++t2) {
      const v8f& dd = t2 ? d1 : d0;
      #pragma unroll
      for (int r = 0; r < 8; r += 2) {
        int il = t2 * 16 + half * 8 + r;   // i_local of rows r, r+1
        float k0 = __expf((dd[r]     - 0.5f * (sq[i0 + il]     + sqj)) * INV_SIGMA2);
        float k1 = __expf((dd[r + 1] - 0.5f * (sq[i0 + il + 1] + sqj)) * INV_SIGMA2);
        union { __bf16 h[2]; unsigned u; } pk;
        pk.h[0] = (__bf16)k0; pk.h[1] = (__bf16)k1;
        *(unsigned*)(kTw + l16 * 32 + il) = pk.u;
      }
    }
    asm volatile("s_wait_dscnt 0" ::: "memory");   // own-wave LDS RAW

    // ---- E[s, j] += Wt_chunk(16x32) x Kbf(32x16), 4 s-blocks ----
    v16bf b2 = ld2(kTw + l16 * 32 + half * 8, kTw + l16 * 32 + 16 + half * 8);
    #pragma unroll
    for (int a = 0; a < 4; ++a) {
      const __bf16* wrow = sWtCc + (a * 16 + l16) * 32;
      v16bf a2 = ld2(wrow + half * 8, wrow + 16 + half * 8);
      accA[a] = __builtin_amdgcn_wmma_f32_16x16x32_bf16(false, a2, false, b2, (short)0, accA[a], false, false);
    }

    // drain own async loads, then barrier: next iteration may read the other
    // buffer written by ALL waves, and may overwrite this one.
    async_wait0();
    __syncthreads();
  }

  // ---- loss partials: l1 = inv_lambda[s] E^2, l2 = E[s,j] W[j,s] ----
  float l1 = 0.0f, l2 = 0.0f;
  #pragma unroll
  for (int a = 0; a < 4; ++a) {
    #pragma unroll
    for (int r = 0; r < 8; ++r) {
      int s = a * 16 + half * 8 + r;
      float e = accA[a][r];
      l1 += invl[s] * e * e;
      l2 += e * W[(size_t)j * S_DIM + s];
    }
  }
  float p = -0.5f * l1 + 0.5f * l2;   // ETA = 1
  #pragma unroll
  for (int o = 16; o > 0; o >>= 1) p += __shfl_xor(p, o, 32);
  if (lane == 0) atomicAdd(accum, p);
}

// ---------------------------------------------------------------------------
// finalize: out = L + (C_STAB/2) L^2
// ---------------------------------------------------------------------------
__global__ void kpca_fin(const float* __restrict__ accum, float* __restrict__ out)
{
  float L = accum[0];
  out[0] = L + 0.05f * L * L;
}

extern "C" void kernel_launch(void* const* d_in, const int* in_sizes, int n_in,
                              void* d_out, int out_size, void* d_ws, size_t ws_size,
                              hipStream_t stream) {
  const float* x    = (const float*)d_in[0];   // [8192][512]
  const float* W    = (const float*)d_in[1];   // [8192][64]
  const float* invl = (const float*)d_in[2];   // [64]
  char* ws = (char*)d_ws;
  __bf16* Xb  = (__bf16*)(ws + XB_OFF);
  __bf16* Wt  = (__bf16*)(ws + WT_OFF);
  float*  sqv = (float*)(ws + SQ_OFF);
  float*  acc = (float*)(ws + ACC_OFF);
  float*  out = (float*)d_out;

  kpca_prep<<<N_SAMPLE, 256, 0, stream>>>(x, W, Xb, Wt, sqv, acc);
  kpca_main<<<N_SAMPLE / 64, 128, L_TOTAL, stream>>>(Xb, Wt, sqv, W, invl, acc);
  kpca_fin<<<1, 1, 0, stream>>>(acc, out);
}
